// QuantizedSparseConv2d_6554120093746
// MI455X (gfx1250) — matrix operand order repaired and checked by reading, not compiled, not run on hardware
//
#include <hip/hip_runtime.h>
#include <math.h>

typedef __attribute__((ext_vector_type(16))) _Float16       v16h;
typedef __attribute__((ext_vector_type(4)))  _Float16       v4h;
typedef __attribute__((ext_vector_type(16))) unsigned short v16u;
typedef __attribute__((ext_vector_type(8)))  float          v8f;

#define IC_   64
#define OC_   64
#define H_    224
#define W_    224
#define HW_   (H_ * W_)
#define B_    16
#define KK_   (IC_ * 9)        // 576 reduction dim
#define KT_   (KK_ / 32)       // 18 K-steps of 32
#define PADW_ 232              // padded LDS row: cols 0..3 and 228..231 are zero pad

// ---------------------------------------------------------------------------
// Async copy of 16B global -> LDS (gfx1250 GLOBAL_LOAD_ASYNC_TO_LDS_B128,
// GVS addressing: saddr base + 32-bit vaddr byte offset). Tracked by ASYNCcnt.
// LDS flat addresses carry the LDS byte offset in the low 32 bits (ISA 10.2).
// ---------------------------------------------------------------------------
__device__ __forceinline__ void async_copy_b128(const void* gbase,
                                                unsigned int gbyte_off,
                                                void* lds_dst) {
  unsigned int lds = (unsigned int)(unsigned long long)lds_dst;
  asm volatile("global_load_async_to_lds_b128 %0, %1, %2"
               :
               : "v"(lds), "v"(gbyte_off), "s"(gbase)
               : "memory");
}
__device__ __forceinline__ void async_wait0() {
  asm volatile("s_wait_asynccnt 0x0" ::: "memory");
}

// ---------------------------------------------------------------------------
// Kernel 1: SAWB statistics over masked weights -> scale (single block)
// ---------------------------------------------------------------------------
__global__ __launch_bounds__(256) void sawb_stats_kernel(
    const float* __restrict__ Wt, const unsigned char* __restrict__ mask,
    float* __restrict__ stats) {
  __shared__ float s_abs[256];
  __shared__ float s_sq[256];
  __shared__ float s_cnt[256];
  const int tid = threadIdx.x;
  float sa = 0.f, sq = 0.f, sc = 0.f;
  for (int i = tid; i < OC_ * KK_; i += 256) {
    float m = mask[i] ? 1.f : 0.f;
    float w = Wt[i] * m;
    sa += fabsf(w);
    sq += w * w;
    sc += m;
  }
  s_abs[tid] = sa; s_sq[tid] = sq; s_cnt[tid] = sc;
  __syncthreads();
  for (int off = 128; off > 0; off >>= 1) {
    if (tid < off) {
      s_abs[tid] += s_abs[tid + off];
      s_sq[tid]  += s_sq[tid + off];
      s_cnt[tid] += s_cnt[tid + off];
    }
    __syncthreads();
  }
  if (tid == 0) {
    float n  = fmaxf(s_cnt[0], 1.f);
    float m1 = s_abs[0] / n;
    float m2 = s_sq[0] / n;
    float alpha = 12.68f * sqrtf(m2) - 12.80f * m1;
    stats[0] = alpha / 127.f;
  }
}

// ---------------------------------------------------------------------------
// Kernel 2: quantize (RNE, clip +-127), dequantize, mask, and pack weights
// into the wmma_f32_16x16x32_f16 A-fragment per-lane layout.
// Fragment (kt, mt) at Apack[(kt*4 + mt)*512 + lane*16 + e].
// ---------------------------------------------------------------------------
__global__ __launch_bounds__(256) void quant_pack_kernel(
    const float* __restrict__ Wt, const unsigned char* __restrict__ mask,
    const float* __restrict__ stats, _Float16* __restrict__ Apack) {
  const int idx = blockIdx.x * 256 + threadIdx.x;
  if (idx >= KT_ * 4 * 512) return;          // 36864 packed elements
  const float scale = stats[0];
  const int kt   = idx >> 11;
  const int rem  = idx & 2047;
  const int mt   = rem >> 9;
  const int lane = (rem >> 4) & 31;
  const int e    = rem & 15;
  const int hi   = lane >> 4;
  const int v    = e >> 1;
  const int kl   = (hi ? 16 : 0) + e + 0 * v;
  const int K    = kt * 32 + kl;
  const int oc   = mt * 16 + (lane & 15);
  const int wi   = oc * KK_ + K;             // (OC, IC, 3, 3) flat == oc*576 + K
  const float m  = mask[wi] ? 1.f : 0.f;
  const float w  = Wt[wi];
  const float q  = rintf(fminf(fmaxf(w / scale, -127.f), 127.f));
  Apack[idx] = (_Float16)(q * scale * m);
}

// ---------------------------------------------------------------------------
// Kernel 3: implicit-GEMM conv, one block per output row (b,h).
// 14 waves x 16 cols = 224 cols; each wave accumulates all 64 OC.
// IC staged in 2 chunks of 32: x halo tile in LDS as zero-padded f16.
// A slices double-buffered in LDS via async-to-LDS copies (ASYNCcnt).
// ---------------------------------------------------------------------------
__global__ __launch_bounds__(448) void conv_wmma_kernel(
    const float* __restrict__ x, const float* __restrict__ bias,
    const _Float16* __restrict__ Apack, float* __restrict__ out) {
  __shared__ __align__(16) _Float16 sX[32 * 3 * PADW_];   // 43.5 KB halo tile
  __shared__ __align__(16) _Float16 sA[2][2048];          // 8 KB A double buffer
  __shared__ unsigned short s_off2[KK_];                  // tap offsets into sX
  __shared__ float s_bias[OC_];

  const int tid  = threadIdx.x;
  const int wave = tid >> 5;
  const int lane = tid & 31;

  // offset table: K -> LDS index of (ic%32, dh, dw) tap at column 0
  for (int K = tid; K < KK_; K += 448) {
    const int ic = K / 9, r = K - ic * 9;
    s_off2[K] = (unsigned short)(((ic & 31) * 3 + r / 3) * PADW_ + (r % 3) + 3);
  }
  if (tid < OC_) s_bias[tid] = bias[tid];

  // kick off async staging of the first A slice (waves 0..7: 256 x 16B = 4KB)
  if (tid < 256)
    async_copy_b128(Apack, (unsigned int)tid * 16u, &sA[0][tid * 8]);

  const int b    = blockIdx.x / H_;
  const int h    = blockIdx.x - b * H_;
  const int w    = (wave << 4) + (lane & 15);   // this lane's output column
  const int hi16 = lane & 16;                   // K sub-range selector

  v8f acc0 = {}, acc1 = {}, acc2 = {}, acc3 = {};

  for (int chunk = 0; chunk < 2; ++chunk) {
    __syncthreads();   // previous chunk's compute done before restaging

    // zero the left/right halo pad columns (cols 0..3 and 228..231)
    for (int i = tid; i < 32 * 3; i += 448) {
      const v4h z = {};
      *(v4h*)&sX[i * PADW_ + 0]   = z;
      *(v4h*)&sX[i * PADW_ + 228] = z;
    }
    // stage 32 ic x 3 rows x 224 cols as f16 (float4 -> v4h, 8B-aligned stores)
    for (int i = tid; i < 32 * 3 * 56; i += 448) {
      const int g     = i % 56;        // float4 group within row
      const int rowid = i / 56;        // (icl*3 + j)
      const int j     = rowid % 3;
      const int icl   = rowid / 3;
      const int hr    = h + j - 1;
      const int c     = g * 4;
      v4h hv = {};
      if ((unsigned)hr < (unsigned)H_) {
        const float4 f = *(const float4*)
            &x[(((b * IC_) + chunk * 32 + icl) * H_ + hr) * W_ + c];
        hv.x = (_Float16)f.x; hv.y = (_Float16)f.y;
        hv.z = (_Float16)f.z; hv.w = (_Float16)f.w;
      }
      *(v4h*)&sX[rowid * PADW_ + 4 + c] = hv;   // data col c lands at pad+4+c
    }

    // 9 K-steps of 32 over this IC chunk
    for (int kt9 = 0; kt9 < 9; ++kt9) {
      const int kt  = chunk * 9 + kt9;
      const int buf = kt & 1;

      if (tid < 256) async_wait0();   // current A slice landed in LDS
      __syncthreads();                // everyone sees sA[buf] (and sX on kt9==0);
                                      // also: no wave still reads sA[buf^1]

      // overlap: stage the next A slice into the other buffer during compute
      if (kt + 1 < KT_ && tid < 256)
        async_copy_b128(Apack, (unsigned int)(kt + 1) * 4096u + (unsigned int)tid * 16u,
                        &sA[(kt + 1) & 1][tid * 8]);

      // 16 consecutive tap offsets in one 32B vector read
      const int  kb   = kt * 32 + hi16;
      const v16u offv = *(const v16u*)&s_off2[kb];
      v16h bf;
#pragma unroll
      for (int e = 0; e < 16; ++e)
        bf[e] = sX[(int)offv[e] + w];

      const v16h a0 = *(const v16h*)&sA[buf][0 * 512 + lane * 16];
      const v16h a1 = *(const v16h*)&sA[buf][1 * 512 + lane * 16];
      const v16h a2 = *(const v16h*)&sA[buf][2 * 512 + lane * 16];
      const v16h a3 = *(const v16h*)&sA[buf][3 * 512 + lane * 16];

      acc0 = __builtin_amdgcn_wmma_f32_16x16x32_f16(false, a0, false, bf,
                                                    (short)0, acc0, false, false);
      acc1 = __builtin_amdgcn_wmma_f32_16x16x32_f16(false, a1, false, bf,
                                                    (short)0, acc1, false, false);
      acc2 = __builtin_amdgcn_wmma_f32_16x16x32_f16(false, a2, false, bf,
                                                    (short)0, acc2, false, false);
      acc3 = __builtin_amdgcn_wmma_f32_16x16x32_f16(false, a3, false, bf,
                                                    (short)0, acc3, false, false);
    }
  }

  // epilogue: C layout row M = r + (lane>=16 ? 8 : 0), col N = lane&15
  const int rbase = (lane & 16) ? 8 : 0;
  const int ob    = b * (OC_ * HW_) + h * W_ + w;
#pragma unroll
  for (int r = 0; r < 8; ++r) {
    const int M = rbase + r;
    out[ob + (0 * 16 + M) * HW_] = acc0[r] + s_bias[0 * 16 + M];
    out[ob + (1 * 16 + M) * HW_] = acc1[r] + s_bias[1 * 16 + M];
    out[ob + (2 * 16 + M) * HW_] = acc2[r] + s_bias[2 * 16 + M];
    out[ob + (3 * 16 + M) * HW_] = acc3[r] + s_bias[3 * 16 + M];
  }
}

// ---------------------------------------------------------------------------
extern "C" void kernel_launch(void* const* d_in, const int* in_sizes, int n_in,
                              void* d_out, int out_size, void* d_ws, size_t ws_size,
                              hipStream_t stream) {
  const float*         x    = (const float*)d_in[0];
  const float*         Wt   = (const float*)d_in[1];
  const unsigned char* mask = (const unsigned char*)d_in[2];  // bool array
  const float*         bias = (const float*)d_in[3];
  float* out = (float*)d_out;

  float*    stats = (float*)d_ws;                           // [0] = scale
  _Float16* Apack = (_Float16*)((char*)d_ws + 64);          // 36864 f16 = 72 KB

  sawb_stats_kernel<<<1, 256, 0, stream>>>(Wt, mask, stats);
  quant_pack_kernel<<<(KT_ * 4 * 512 + 255) / 256, 256, 0, stream>>>(
      Wt, mask, stats, Apack);
  conv_wmma_kernel<<<B_ * H_, 448, 0, stream>>>(x, bias, Apack, out);
}
